// DendriticFractalBlock_80994493268381
// MI455X (gfx1250) — matrix operand order repaired and checked by reading, not compiled, not run on hardware
//
#include <hip/hip_runtime.h>
#include <hip/hip_bf16.h>

typedef __attribute__((ext_vector_type(16))) _Float16 v16h;
typedef __attribute__((ext_vector_type(8)))  float    v8f;

#define TOK  4096
#define SEQ  2048
#define DIM  1024
#define O3   3072
#define FFD  2048
#define NDEN 4
#define DD   256
#define HEADS 16
#define HD   64

// ---------------------------------------------------------------- WMMA helper
__device__ __forceinline__ v8f wmma16(v16h a, v16h b, v8f c) {
  return __builtin_amdgcn_wmma_f32_16x16x32_f16(
      /*neg_a=*/false, a, /*neg_b=*/false, b,
      /*c_mod=*/(short)0, c, /*reuse_a=*/false, /*reuse_b=*/false);
}

__device__ __forceinline__ float fast_rcp(float x) {
  return __builtin_amdgcn_rcpf(x);
}
__device__ __forceinline__ float silu(float s) {
  return s * fast_rcp(1.f + __expf(-s));
}

// A fragment: 16xK tile at p = &A[row0][k0], row stride lda (elements).
// Contiguous K order used here (not the ISA table order): lanes 0-15 -> rows, K 0..15; lanes 16-31 -> K 16..31
__device__ __forceinline__ v16h load_a_frag(const _Float16* __restrict__ p,
                                            int lda, int lane) {
  int row = lane & 15;
  int grp = lane >> 4;
  const _Float16* q = p + (size_t)row * lda + grp * 16;
  v16h a;
#pragma unroll
  for (int i = 0; i < 8; ++i) { a[i] = q[i]; a[8 + i] = q[8 + i]; }
  return a;
}

// A fragment from f16 LDS tile (same layout, ds_load path)
__device__ __forceinline__ v16h load_a_frag_ldsh(const _Float16* s,
                                                 int lda, int lane) {
  int row = lane & 15;
  int grp = lane >> 4;
  const _Float16* q = s + row * lda + grp * 16;
  v16h a;
#pragma unroll
  for (int i = 0; i < 8; ++i) { a[i] = q[i]; a[8 + i] = q[8 + i]; }
  return a;
}

// A fragment sourced from f32 LDS tile (converted to f16 on the fly)
__device__ __forceinline__ v16h load_a_frag_lds(const float* __restrict__ s,
                                                int lds, int lane) {
  int row = lane & 15;
  int grp = lane >> 4;
  const float* q = s + row * lds + grp * 16;
  v16h a;
#pragma unroll
  for (int i = 0; i < 8; ++i) {
    a[i]     = (_Float16)q[i];
    a[8 + i] = (_Float16)q[8 + i];
  }
  return a;
}

// B fragment, K contiguous in memory: column stride ldb.
// lanes 0-15: col = lane, K 0..15 ; lanes 16-31: col = lane-16, K 16..31
__device__ __forceinline__ v16h load_b_frag_kc(const _Float16* __restrict__ p,
                                               int ldb, int lane) {
  int col = lane & 15;
  int grp = lane >> 4;
  const _Float16* q = p + (size_t)col * ldb + grp * 16;
  v16h b;
#pragma unroll
  for (int i = 0; i < 16; ++i) b[i] = q[i];
  return b;
}

// B fragment, K strided (stride ks between K rows), columns contiguous.
__device__ __forceinline__ v16h load_b_frag_ks(const _Float16* __restrict__ p,
                                               int ks, int lane) {
  int col = lane & 15;
  int grp = lane >> 4;
  const _Float16* q = p + col + (size_t)grp * 16 * ks;
  v16h b;
#pragma unroll
  for (int i = 0; i < 16; ++i) b[i] = q[(size_t)i * ks];
  return b;
}

// ------------------------------------------------------- weight converters
__global__ void k_tconv_dendritic(const float* __restrict__ w,   // (ND, DD, O)
                                  _Float16* __restrict__ wt,     // (ND, O, DD)
                                  int O) {
  int idx = blockIdx.x * blockDim.x + threadIdx.x;
  int total = NDEN * DD * O;
  if (idx >= total) return;
  int o = idx % O;
  int t = idx / O;
  int d = t % DD;
  int n = t / DD;
  wt[((size_t)n * O + o) * DD + d] = (_Float16)w[idx];
}

__global__ void k_tconv_2d(const float* __restrict__ w,  // (K, N)
                           _Float16* __restrict__ wt,    // (N, K)
                           int K, int N) {
  int idx = blockIdx.x * blockDim.x + threadIdx.x;
  if (idx >= K * N) return;
  int n = idx % N;
  int k = idx / N;
  wt[(size_t)n * K + k] = (_Float16)w[idx];
}

// ---------------------------------------------------------------- RMSNorm
__global__ void k_rmsnorm(const float* __restrict__ x,
                          const float* __restrict__ w,
                          _Float16* __restrict__ out) {
  __shared__ float red[8];
  int row = blockIdx.x;
  const float* xr = x + (size_t)row * DIM;
  float ss = 0.f;
  for (int i = threadIdx.x; i < DIM; i += blockDim.x) {
    float v = xr[i];
    ss += v * v;
  }
#pragma unroll
  for (int o = 16; o; o >>= 1) ss += __shfl_down(ss, o, 32);
  if ((threadIdx.x & 31) == 0) red[threadIdx.x >> 5] = ss;
  __syncthreads();
  if (threadIdx.x == 0) {
    float t = 0.f;
    for (int i = 0; i < 8; ++i) t += red[i];
    red[0] = rsqrtf(t / (float)DIM + 1.1920929e-7f);
  }
  __syncthreads();
  float r = red[0];
  for (int i = threadIdx.x; i < DIM; i += blockDim.x)
    out[(size_t)row * DIM + i] = (_Float16)(xr[i] * r * w[i]);
}

// --------------------------------------------- dendritic branch-max GEMM
// Tile per wave: 16 rows x 32 cols (A fragment reused across 2 B tiles).
// out[t,o] = max_n gain[n,o]*silu(sum_d h[t,n*DD+d]*w[n,d,o] - thr[n,o])
__global__ void k_dendritic(const _Float16* __restrict__ h,   // TOK x DIM f16
                            const _Float16* __restrict__ wt,  // ND x O x DD
                            const float* __restrict__ thr,    // ND x O
                            const float* __restrict__ gain,   // ND x O
                            _Float16* __restrict__ out,       // TOK x O
                            int O, int extra_silu) {
  int lane = threadIdx.x;
  int col0 = blockIdx.x * 32;
  int row0 = blockIdx.y * 16;
  float best[2][8];
#pragma unroll
  for (int ct = 0; ct < 2; ++ct)
#pragma unroll
    for (int v = 0; v < 8; ++v) best[ct][v] = -3.4e38f;

  for (int n = 0; n < NDEN; ++n) {
    v8f acc[2] = {};
    const _Float16* ab = h + (size_t)row0 * DIM + n * DD;
    const _Float16* bb = wt + ((size_t)n * O + col0) * DD;
#pragma unroll 2
    for (int k = 0; k < DD; k += 32) {
      __builtin_prefetch(bb + k + 128, 0, 0);
      v16h a = load_a_frag(ab + k, DIM, lane);
      v16h b0 = load_b_frag_kc(bb + k, DD, lane);
      v16h b1 = load_b_frag_kc(bb + 16 * DD + k, DD, lane);
      acc[0] = wmma16(a, b0, acc[0]);
      acc[1] = wmma16(a, b1, acc[1]);
    }
#pragma unroll
    for (int ct = 0; ct < 2; ++ct) {
      int col = col0 + ct * 16 + (lane & 15);
      float tv = thr[(size_t)n * O + col];
      float gv = gain[(size_t)n * O + col];
#pragma unroll
      for (int v = 0; v < 8; ++v) {
        float s = acc[ct][v] - tv;
        best[ct][v] = fmaxf(best[ct][v], gv * silu(s));
      }
    }
  }
  int rb = row0 + ((lane >> 4) ? 8 : 0);
#pragma unroll
  for (int ct = 0; ct < 2; ++ct) {
    int col = col0 + ct * 16 + (lane & 15);
#pragma unroll
    for (int v = 0; v < 8; ++v) {
      float val = best[ct][v];
      if (extra_silu) val = silu(val);
      out[(size_t)(rb + v) * O + col] = (_Float16)val;
    }
  }
}

// ------------------------------------------------ flash attention (causal)
// qkv: TOK x 3072 f16 (q|k|v, head-major within each 1024 chunk)
__global__ void k_attention(const _Float16* __restrict__ qkv,
                            _Float16* __restrict__ outp) {   // TOK x DIM f16
  int lane = threadIdx.x;
  int q0 = blockIdx.x * 16;
  int bh = blockIdx.y;
  int b = bh >> 4, hh = bh & 15;
  const _Float16* qb = qkv + (size_t)b * SEQ * O3 + hh * HD;
  const _Float16* kb = qb + DIM;
  const _Float16* vb = qb + 2 * DIM;

  __shared__ float sP[16 * 32];
  __shared__ float sF[16];
  __shared__ float sM[16];
  __shared__ float sL[16];
  if (lane < 16) { sM[lane] = -3.4e38f; sL[lane] = 0.f; }
  __syncthreads();

  v8f acc[4] = {};
  const float scale = 0.125f;  // 1/sqrt(64)
  const _Float16* ab = qb + (size_t)q0 * O3;

  for (int kb0 = 0; kb0 < q0 + 16; kb0 += 32) {
    // ---- S = Q K^T (16x32), two 16x16 tiles, K(hd)=64 in two steps
#pragma unroll
    for (int nc = 0; nc < 2; ++nc) {
      const _Float16* bb = kb + (size_t)(kb0 + nc * 16) * O3;
      v8f s = {};
      s = wmma16(load_a_frag(ab, O3, lane),      load_b_frag_kc(bb, O3, lane),      s);
      s = wmma16(load_a_frag(ab + 32, O3, lane), load_b_frag_kc(bb + 32, O3, lane), s);
      int c = lane & 15;
      int key = kb0 + nc * 16 + c;
      int rb = (lane >> 4) ? 8 : 0;
#pragma unroll
      for (int v = 0; v < 8; ++v) {
        int qi = q0 + rb + v;
        sP[(rb + v) * 32 + nc * 16 + c] = (key <= qi) ? s[v] * scale : -3.4e38f;
      }
    }
    __syncthreads();
    // ---- online softmax for this key block (lane r owns query row r)
    if (lane < 16) {
      float m_old = sM[lane];
      float mx = m_old;
#pragma unroll 8
      for (int j = 0; j < 32; ++j) mx = fmaxf(mx, sP[lane * 32 + j]);
      float f = __expf(m_old - mx);
      float sum = 0.f;
#pragma unroll 8
      for (int j = 0; j < 32; ++j) {
        float p = __expf(sP[lane * 32 + j] - mx);
        sP[lane * 32 + j] = p;
        sum += p;
      }
      sM[lane] = mx;
      sL[lane] = sL[lane] * f + sum;
      sF[lane] = f;
    }
    __syncthreads();
    // ---- O = O*f + P V  (P re-read from LDS as f16 A fragment)
    int rb = (lane >> 4) ? 8 : 0;
    float fr[8];
#pragma unroll
    for (int v = 0; v < 8; ++v) fr[v] = sF[rb + v];
    v16h pfrag = load_a_frag_lds(sP, 32, lane);
#pragma unroll
    for (int dt = 0; dt < 4; ++dt) {
#pragma unroll
      for (int v = 0; v < 8; ++v) acc[dt][v] *= fr[v];
      v16h vfrag = load_b_frag_ks(vb + (size_t)kb0 * O3 + dt * 16, O3, lane);
      acc[dt] = wmma16(pfrag, vfrag, acc[dt]);
    }
    __syncthreads();
  }
  // ---- epilogue: divide by l (v_rcp), store head-major into TOK x DIM
  int rb = (lane >> 4) ? 8 : 0;
  int c = lane & 15;
  float linv[8];
#pragma unroll
  for (int v = 0; v < 8; ++v) linv[v] = fast_rcp(sL[rb + v]);
#pragma unroll
  for (int dt = 0; dt < 4; ++dt)
#pragma unroll
    for (int v = 0; v < 8; ++v)
      outp[(size_t)(b * SEQ + q0 + rb + v) * DIM + hh * HD + dt * 16 + c] =
          (_Float16)(acc[dt][v] * linv[v]);
}

// -------------------------------------------- GEMM + residual (f32 out)
// Tile per wave: 16 rows x 64 cols. The 16xK A tile is staged into LDS once
// via the CDNA5 async-to-LDS engine (ASYNCcnt), then WMMA A fragments are
// sourced from LDS while B streams from L2.
template <int K>
__global__ void k_gemm_residual(const _Float16* __restrict__ act, // TOK x K
                                const _Float16* __restrict__ wt,  // N x K
                                const float* __restrict__ res,    // TOK x N
                                float* __restrict__ out,          // TOK x N
                                int N) {
  extern __shared__ char smem_raw[];
  _Float16* sA = (_Float16*)smem_raw;        // 16 x K f16 tile
  int lane = threadIdx.x;
  int col0 = blockIdx.x * 64;
  int row0 = blockIdx.y * 16;

  // ---- async stage A tile (row-major, 16B chunks, 512B per instruction)
  constexpr int CPR = K / 8;                 // 16B chunks per row
  constexpr int NCH = 2 * K;                 // total 16B chunks (16 rows)
#pragma unroll 4
  for (int c = lane; c < NCH; c += 32) {
    int row = c / CPR;                       // shift (K is constexpr pow2)
    int off = c % CPR;
    unsigned long long ga =
        (unsigned long long)(act + ((size_t)(row0 + row) * K + off * 8));
    unsigned lds_off = (unsigned)c * 16u;
    asm volatile("global_load_async_to_lds_b128 %0, %1, off"
                 :: "v"(lds_off), "v"(ga) : "memory");
  }
  asm volatile("s_wait_asynccnt 0x0" ::: "memory");
  __syncthreads();

  v8f acc[4] = {};
  const _Float16* bb = wt + (size_t)col0 * K;
#pragma unroll 2
  for (int k = 0; k < K; k += 32) {
    v16h a = load_a_frag_ldsh(sA + k, K, lane);
#pragma unroll
    for (int ct = 0; ct < 4; ++ct) {
      __builtin_prefetch(bb + (size_t)ct * 16 * K + k + 256, 0, 0);
      v16h b = load_b_frag_kc(bb + (size_t)ct * 16 * K + k, K, lane);
      acc[ct] = wmma16(a, b, acc[ct]);
    }
  }
  int rb = row0 + ((lane >> 4) ? 8 : 0);
#pragma unroll
  for (int ct = 0; ct < 4; ++ct) {
    int c = col0 + ct * 16 + (lane & 15);
#pragma unroll
    for (int v = 0; v < 8; ++v)
      out[(size_t)(rb + v) * N + c] = res[(size_t)(rb + v) * N + c] + acc[ct][v];
  }
}

// ------------------------------------------------------- elementwise mul
__global__ void k_mul(const _Float16* __restrict__ a,
                      const _Float16* __restrict__ b,
                      _Float16* __restrict__ o, int n) {
  int i = blockIdx.x * blockDim.x + threadIdx.x;
  if (i < n) o[i] = (_Float16)((float)a[i] * (float)b[i]);
}

// ================================================================ launch
extern "C" void kernel_launch(void* const* d_in, const int* in_sizes, int n_in,
                              void* d_out, int out_size, void* d_ws, size_t ws_size,
                              hipStream_t stream) {
  const float* x        = (const float*)d_in[0];
  const float* qkv_w    = (const float*)d_in[1];
  const float* qkv_thr  = (const float*)d_in[2];
  const float* qkv_gain = (const float*)d_in[3];
  const float* o_w      = (const float*)d_in[4];
  const float* gate_w   = (const float*)d_in[5];
  const float* gate_thr = (const float*)d_in[6];
  const float* gate_gain= (const float*)d_in[7];
  const float* up_w     = (const float*)d_in[8];
  const float* up_thr   = (const float*)d_in[9];
  const float* up_gain  = (const float*)d_in[10];
  const float* down_w   = (const float*)d_in[11];
  const float* norm1_w  = (const float*)d_in[12];
  const float* norm2_w  = (const float*)d_in[13];
  float* out = (float*)d_out;

  char* ws = (char*)d_ws;
  size_t off = 0;
  auto alloc = [&](size_t bytes) -> void* {
    void* p = ws + off;
    off = (off + bytes + 255) & ~(size_t)255;
    return p;
  };
  _Float16* h1      = (_Float16*)alloc((size_t)TOK * DIM * 2);
  _Float16* qkv_wt  = (_Float16*)alloc((size_t)NDEN * O3 * DD * 2);
  _Float16* qkv_out = (_Float16*)alloc((size_t)TOK * O3 * 2);
  _Float16* attn_o  = (_Float16*)alloc((size_t)TOK * DIM * 2);
  _Float16* o_wt    = (_Float16*)alloc((size_t)DIM * DIM * 2);
  float*    x2      = (float*)   alloc((size_t)TOK * DIM * 4);
  _Float16* h2      = (_Float16*)alloc((size_t)TOK * DIM * 2);
  _Float16* gate_wt = (_Float16*)alloc((size_t)NDEN * FFD * DD * 2);
  _Float16* up_wt   = (_Float16*)alloc((size_t)NDEN * FFD * DD * 2);
  _Float16* g_out   = (_Float16*)alloc((size_t)TOK * FFD * 2);
  _Float16* u_out   = (_Float16*)alloc((size_t)TOK * FFD * 2);
  _Float16* ff      = (_Float16*)alloc((size_t)TOK * FFD * 2);
  _Float16* down_wt = (_Float16*)alloc((size_t)DIM * FFD * 2);

  // ---- weight conversion / transposition (f32 -> f16, K-contiguous)
  {
    int t1 = NDEN * DD * O3;
    k_tconv_dendritic<<<(t1 + 255) / 256, 256, 0, stream>>>(qkv_w, qkv_wt, O3);
    int t2 = NDEN * DD * FFD;
    k_tconv_dendritic<<<(t2 + 255) / 256, 256, 0, stream>>>(gate_w, gate_wt, FFD);
    k_tconv_dendritic<<<(t2 + 255) / 256, 256, 0, stream>>>(up_w, up_wt, FFD);
    k_tconv_2d<<<(DIM * DIM + 255) / 256, 256, 0, stream>>>(o_w, o_wt, DIM, DIM);
    k_tconv_2d<<<(FFD * DIM + 255) / 256, 256, 0, stream>>>(down_w, down_wt, FFD, DIM);
  }

  // ---- norm1 -> h1 (f16)
  k_rmsnorm<<<TOK, 256, 0, stream>>>(x, norm1_w, h1);

  // ---- dendritic QKV: TOK x 3072
  k_dendritic<<<dim3(O3 / 32, TOK / 16), 32, 0, stream>>>(
      h1, qkv_wt, qkv_thr, qkv_gain, qkv_out, O3, 0);

  // ---- causal flash attention per (b, h, 16-query tile)
  k_attention<<<dim3(SEQ / 16, 2 * HEADS), 32, 0, stream>>>(qkv_out, attn_o);

  // ---- x2 = x + attn_o @ o_w   (A tile async-staged to LDS)
  k_gemm_residual<DIM><<<dim3(DIM / 64, TOK / 16), 32, 16 * DIM * 2, stream>>>(
      attn_o, o_wt, x, x2, DIM);

  // ---- norm2 -> h2
  k_rmsnorm<<<TOK, 256, 0, stream>>>(x2, norm2_w, h2);

  // ---- dendritic gate (with extra silu) and up
  k_dendritic<<<dim3(FFD / 32, TOK / 16), 32, 0, stream>>>(
      h2, gate_wt, gate_thr, gate_gain, g_out, FFD, 1);
  k_dendritic<<<dim3(FFD / 32, TOK / 16), 32, 0, stream>>>(
      h2, up_wt, up_thr, up_gain, u_out, FFD, 0);

  // ---- ff = silu(gate) * up
  k_mul<<<(TOK * FFD + 255) / 256, 256, 0, stream>>>(g_out, u_out, ff, TOK * FFD);

  // ---- out = x2 + ff @ down_w  (A tile async-staged to LDS)
  k_gemm_residual<FFD><<<dim3(DIM / 64, TOK / 16), 32, 16 * FFD * 2, stream>>>(
      ff, down_wt, x2, out, DIM);
}
